// BahdanauAttentionDecoder_51427938402528
// MI455X (gfx1250) — matrix operand order repaired and checked by reading, not compiled, not run on hardware
//
#include <hip/hip_runtime.h>
#include <hip/hip_bf16.h>
#include <math.h>

// ---------------------------------------------------------------------------
// Bahdanau attention decoder, greedy, T=48 steps, B=32, H=512, V=32000.
// All GEMMs run on v_wmma_f32_16x16x32_bf16 (bf16 weights, f32 accumulate).
// Weights are converted to bf16 once into d_ws (all fit in 192MB L2).
// GEMM K-loop unrolled x2 with two named tile-buffer sets (no register
// rotation): loads of one set overlap WMMAs on the other.
// Requires K % 64 == 0 (holds: K is 512 or 1024 everywhere here).
// ---------------------------------------------------------------------------

#define HID   512
#define TLEN  48
#define BATCH 32
#define VOCAB 32000
#define H2    1024   // 2*HID (GRU input)
#define H3    1536   // 3*HID (GRU gates)

typedef __attribute__((ext_vector_type(16))) __bf16 v16bf;
typedef __attribute__((ext_vector_type(8)))  float  v8f;

union BfTile { v16bf v; unsigned u[8]; };

__device__ __forceinline__ unsigned short f2bf(float f) {
  unsigned u = __float_as_uint(f);
  unsigned rnd = 0x7FFFu + ((u >> 16) & 1u);   // round-to-nearest-even
  return (unsigned short)((u + rnd) >> 16);
}

// A-matrix 16x32 bf16 tile (ISA 7.12.2 "16-bit A-Matrix 16x32"):
// lane row = m0+(lane&15); hi=lane>>4; VGPR p<4: K=2p+w+8*hi ; p>=4: K=16+2(p-4)+w+8*hi
// Byte pattern per lane: 16B at +0 and 16B at +32 (compiler fuses to b128 pairs).
__device__ __forceinline__ v16bf load_tileA(const unsigned short* base, int m0,
                                            int ld, int k0, int lane) {
  BfTile t;
  int m  = m0 + (lane & 15);
  int hi = (lane >> 4) & 1;
  const unsigned short* p = base + (size_t)m * ld + k0 + hi * 8;
#pragma unroll
  for (int i = 0; i < 8; ++i) {
    int kb = (i < 4) ? (2 * i) : (16 + 2 * (i - 4));
    t.u[i] = *reinterpret_cast<const unsigned*>(p + kb);
  }
  return t.v;
}

// B-matrix 32x16 bf16 tile (column n per lane; same K order as A:
// VGPR p<4: K=2p+w+8*hi ; p>=4: K=16+2(p-4)+w+8*hi).
// Weight matrices are [N,K] row-major, so column n of B == weight row n.
__device__ __forceinline__ v16bf load_tileB(const unsigned short* base, int n0,
                                            int ld, int k0, int lane) {
  BfTile t;
  int n  = n0 + (lane & 15);
  int hi = (lane >> 4) & 1;
  const unsigned short* p = base + (size_t)n * ld + k0 + hi * 8;
#pragma unroll
  for (int i = 0; i < 8; ++i)
    t.u[i] = *reinterpret_cast<const unsigned*>(p + ((i < 4) ? (2 * i) : (16 + 2 * (i - 4))));
  return t.v;
}

// C = A[MxK] * B[NxK]^T + bias ; one wave per (32-row, 16-col) output tile.
// B tile loaded once per k-step, reused by both m-tiles.
__global__ void wmma_gemm_bf16(const unsigned short* __restrict__ A,
                               const unsigned short* __restrict__ B,
                               float* __restrict__ C,
                               const float* __restrict__ bias,
                               int M, int N, int K, int ldc) {
  int wave = (blockIdx.x * blockDim.x + threadIdx.x) >> 5;
  int lane = threadIdx.x & 31;
  int ntiles = N >> 4;
  int mgrp   = M >> 5;                 // groups of 32 rows
  if (wave >= ntiles * mgrp) return;   // wave-uniform exit (EXEC stays all-1s)
  int nt = wave % ntiles, mt = wave / ntiles;
  int n0 = nt << 4, m0 = mt << 5;
  int nrow = n0 + (lane & 15);

  v8f acc0 = {};
  v8f acc1 = {};
  // prologue: buffer set X holds tiles for k=0
  v16bf a0x = load_tileA(A, m0,      K, 0, lane);
  v16bf a1x = load_tileA(A, m0 + 16, K, 0, lane);
  v16bf bx  = load_tileB(B, n0,      K, 0, lane);

  for (int k0 = 0; k0 < K; k0 += 64) {
    // buffer set Y: tiles for k0+32 (always valid: K % 64 == 0)
    v16bf a0y = load_tileA(A, m0,      K, k0 + 32, lane);
    v16bf a1y = load_tileA(A, m0 + 16, K, k0 + 32, lane);
    v16bf by  = load_tileB(B, n0,      K, k0 + 32, lane);
    // WMMAs on set X while set Y loads are in flight
    acc0 = __builtin_amdgcn_wmma_f32_16x16x32_bf16(false, a0x, false, bx,
                                                   (short)0, acc0, false, false);
    acc1 = __builtin_amdgcn_wmma_f32_16x16x32_bf16(false, a1x, false, bx,
                                                   (short)0, acc1, false, false);
    if (k0 + 64 < K) {
      // stage next block and refill set X for k0+64
      __builtin_prefetch(B + (size_t)nrow * K + k0 + 96, 0, 1);
      a0x = load_tileA(A, m0,      K, k0 + 64, lane);
      a1x = load_tileA(A, m0 + 16, K, k0 + 64, lane);
      bx  = load_tileB(B, n0,      K, k0 + 64, lane);
    }
    // WMMAs on set Y while set X loads are in flight
    acc0 = __builtin_amdgcn_wmma_f32_16x16x32_bf16(false, a0y, false, by,
                                                   (short)0, acc0, false, false);
    acc1 = __builtin_amdgcn_wmma_f32_16x16x32_bf16(false, a1y, false, by,
                                                   (short)0, acc1, false, false);
  }

  // C layout: lanes 0-15 -> rows r, lanes 16-31 -> rows r+8; col = n0+(lane&15)
  int hi = (lane >> 4) & 1;
  float bv = bias ? bias[nrow] : 0.0f;
#pragma unroll
  for (int r = 0; r < 8; ++r) {
    C[(size_t)(m0 + r + hi * 8)      * ldc + nrow] = acc0[r] + bv;
    C[(size_t)(m0 + 16 + r + hi * 8) * ldc + nrow] = acc1[r] + bv;
  }
}

// ---------------------------------------------------------------------------
__global__ void convert_bf16_kernel(const float* __restrict__ src,
                                    unsigned short* __restrict__ dst, int n) {
  int i = blockIdx.x * blockDim.x + threadIdx.x;
  if (i < n) dst[i] = f2bf(src[i]);
}

// h_s [T,B,H] fp32 -> inputs_bf16 [B,T,H]
__global__ void transpose_convert_kernel(const float* __restrict__ h_s,
                                         unsigned short* __restrict__ dst) {
  int i = blockIdx.x * blockDim.x + threadIdx.x;
  if (i >= TLEN * BATCH * HID) return;
  int h = i & (HID - 1);
  int b = (i >> 9) & (BATCH - 1);
  int t = i >> 14;
  dst[((size_t)b * TLEN + t) * HID + h] = f2bf(h_s[i]);
}

__global__ void init_state_kernel(float* h, unsigned short* hb, int* tok) {
  int i = blockIdx.x * blockDim.x + threadIdx.x;
  if (i < BATCH * HID) { h[i] = 0.0f; hb[i] = 0; }
  if (i < BATCH) tok[i] = 0;
}

// e = Wv . tanh(q + hs_proj) + bv ; softmax over T ; ctx = sum alpha*inputs
__global__ void attention_kernel(const float* __restrict__ q,
                                 const float* __restrict__ hs_proj,
                                 const float* __restrict__ h_s,
                                 const float* __restrict__ Wv,
                                 const float* __restrict__ bv,
                                 float* __restrict__ ctx) {
  int b = blockIdx.x;
  __shared__ float sq[HID];
  __shared__ float se[TLEN];
  __shared__ float sa[TLEN];
  for (int h = threadIdx.x; h < HID; h += blockDim.x) sq[h] = q[b * HID + h];
  __syncthreads();

  int wave = threadIdx.x >> 5, lane = threadIdx.x & 31, nw = blockDim.x >> 5;
  for (int t = wave; t < TLEN; t += nw) {
    const float* hp = hs_proj + ((size_t)b * TLEN + t) * HID;
    float s = 0.0f;
    for (int h = lane; h < HID; h += 32) s += Wv[h] * tanhf(sq[h] + hp[h]);
#pragma unroll
    for (int off = 16; off > 0; off >>= 1) s += __shfl_xor(s, off, 32);
    if (lane == 0) se[t] = s + bv[0];
  }
  __syncthreads();
  if (threadIdx.x == 0) {
    float m = se[0];
    for (int t = 1; t < TLEN; ++t) m = fmaxf(m, se[t]);
    float sum = 0.0f;
    for (int t = 0; t < TLEN; ++t) { float e = __expf(se[t] - m); sa[t] = e; sum += e; }
    float inv = 1.0f / sum;
    for (int t = 0; t < TLEN; ++t) sa[t] *= inv;
  }
  __syncthreads();
  for (int h = threadIdx.x; h < HID; h += blockDim.x) {
    float acc = 0.0f;
    for (int t = 0; t < TLEN; ++t)
      acc += sa[t] * h_s[((size_t)t * BATCH + b) * HID + h];
    ctx[b * HID + h] = acc;
  }
}

// x = [ctx ; emb[tok]] as bf16 [B, 2H]
__global__ void build_x_kernel(const float* __restrict__ ctx,
                               const float* __restrict__ emb,
                               const int* __restrict__ tok,
                               unsigned short* __restrict__ x) {
  int b = blockIdx.x;
  int tk = tok[b];
  for (int h = threadIdx.x; h < HID; h += blockDim.x) {
    x[(size_t)b * H2 + h]       = f2bf(ctx[b * HID + h]);
    x[(size_t)b * H2 + HID + h] = f2bf(emb[(size_t)tk * HID + h]);
  }
}

__global__ void gru_gates_kernel(const float* __restrict__ gi,
                                 const float* __restrict__ gh,
                                 float* __restrict__ h,
                                 unsigned short* __restrict__ hb) {
  int idx = blockIdx.x * blockDim.x + threadIdx.x;
  if (idx >= BATCH * HID) return;
  int b = idx >> 9, j = idx & (HID - 1);
  const float* gib = gi + (size_t)b * H3;
  const float* ghb = gh + (size_t)b * H3;
  float r = 1.0f / (1.0f + __expf(-(gib[j] + ghb[j])));
  float z = 1.0f / (1.0f + __expf(-(gib[HID + j] + ghb[HID + j])));
  float n = tanhf(gib[2 * HID + j] + r * ghb[2 * HID + j]);
  float hn = (1.0f - z) * n + z * h[idx];
  h[idx]  = hn;
  hb[idx] = f2bf(hn);
}

__global__ void argmax_kernel(const float* __restrict__ logits,
                              int* __restrict__ tok) {
  int b = blockIdx.x;
  const float* row = logits + (size_t)b * VOCAB;
  float best = -INFINITY; int bidx = 0;
  for (int v = threadIdx.x; v < VOCAB; v += blockDim.x) {
    float f = row[v];
    if (f > best) { best = f; bidx = v; }
  }
  __shared__ float sv[256];
  __shared__ int   si[256];
  sv[threadIdx.x] = best; si[threadIdx.x] = bidx;
  __syncthreads();
  for (int s = blockDim.x >> 1; s > 0; s >>= 1) {
    if ((int)threadIdx.x < s) {
      float ov = sv[threadIdx.x + s]; int oi = si[threadIdx.x + s];
      if (ov > sv[threadIdx.x] ||
          (ov == sv[threadIdx.x] && oi < si[threadIdx.x])) {
        sv[threadIdx.x] = ov; si[threadIdx.x] = oi;
      }
    }
    __syncthreads();
  }
  if (threadIdx.x == 0) tok[b] = si[0];
}

// ---------------------------------------------------------------------------
extern "C" void kernel_launch(void* const* d_in, const int* in_sizes, int n_in,
                              void* d_out, int out_size, void* d_ws, size_t ws_size,
                              hipStream_t stream) {
  (void)in_sizes; (void)n_in; (void)out_size; (void)ws_size;
  const float* h_s  = (const float*)d_in[0];
  const float* Wi   = (const float*)d_in[1];
  const float* bi   = (const float*)d_in[2];
  const float* Wh   = (const float*)d_in[3];
  const float* bh   = (const float*)d_in[4];
  const float* Wv   = (const float*)d_in[5];
  const float* bv   = (const float*)d_in[6];
  const float* emb  = (const float*)d_in[7];
  const float* W_ih = (const float*)d_in[8];
  const float* b_ih = (const float*)d_in[9];
  const float* W_hh = (const float*)d_in[10];
  const float* b_hh = (const float*)d_in[11];
  const float* Wout = (const float*)d_in[12];
  const float* bout = (const float*)d_in[13];
  float* out = (float*)d_out;

  // workspace carve-up (256B aligned)
  char* ws = (char*)d_ws;
  size_t off = 0;
  auto carve = [&](size_t bytes) -> char* {
    char* p = ws + off;
    off = (off + bytes + 255) & ~(size_t)255;
    return p;
  };
  unsigned short* Wout_b = (unsigned short*)carve((size_t)VOCAB * HID * 2);
  unsigned short* Wih_b  = (unsigned short*)carve((size_t)H3 * H2 * 2);
  unsigned short* Whh_b  = (unsigned short*)carve((size_t)H3 * HID * 2);
  unsigned short* Wh_b   = (unsigned short*)carve((size_t)HID * HID * 2);
  unsigned short* Wi_b   = (unsigned short*)carve((size_t)HID * HID * 2);
  unsigned short* inp_b  = (unsigned short*)carve((size_t)BATCH * TLEN * HID * 2);
  float*          hsproj = (float*)carve((size_t)BATCH * TLEN * HID * 4);
  float*          hst    = (float*)carve((size_t)BATCH * HID * 4);
  unsigned short* hst_b  = (unsigned short*)carve((size_t)BATCH * HID * 2);
  float*          qb     = (float*)carve((size_t)BATCH * HID * 4);
  float*          ctx    = (float*)carve((size_t)BATCH * HID * 4);
  unsigned short* xb     = (unsigned short*)carve((size_t)BATCH * H2 * 2);
  float*          gi     = (float*)carve((size_t)BATCH * H3 * 4);
  float*          gh     = (float*)carve((size_t)BATCH * H3 * 4);
  int*            tok    = (int*)carve((size_t)BATCH * 4);

  auto cvt = [&](const float* s, unsigned short* d, size_t n) {
    convert_bf16_kernel<<<(unsigned)((n + 255) / 256), 256, 0, stream>>>(s, d, (int)n);
  };
  auto gemm = [&](const unsigned short* A, const unsigned short* B, float* C,
                  const float* bias, int M, int N, int K, int ldc) {
    int waves = (N >> 4) * (M >> 5);
    int grid  = (waves * 32 + 255) / 256;
    wmma_gemm_bf16<<<grid, 256, 0, stream>>>(A, B, C, bias, M, N, K, ldc);
  };

  // ---- one-time setup (weights stay L2-resident: ~38MB bf16 << 192MB L2) ----
  cvt(Wout, Wout_b, (size_t)VOCAB * HID);
  cvt(W_ih, Wih_b,  (size_t)H3 * H2);
  cvt(W_hh, Whh_b,  (size_t)H3 * HID);
  cvt(Wh,   Wh_b,   (size_t)HID * HID);
  cvt(Wi,   Wi_b,   (size_t)HID * HID);
  transpose_convert_kernel<<<(TLEN * BATCH * HID + 255) / 256, 256, 0, stream>>>(h_s, inp_b);
  init_state_kernel<<<(BATCH * HID + 255) / 256, 256, 0, stream>>>(hst, hst_b, tok);
  // hs_proj = inputs @ Wi^T + bi   [B*T, H]
  gemm(inp_b, Wi_b, hsproj, bi, BATCH * TLEN, HID, HID, HID);

  // ---- 48 sequential greedy decode steps ----
  for (int t = 0; t < TLEN; ++t) {
    float* logits = out + (size_t)t * BATCH * VOCAB;
    // q = h @ Wh^T + bh
    gemm(hst_b, Wh_b, qb, bh, BATCH, HID, HID, HID);
    attention_kernel<<<BATCH, 256, 0, stream>>>(qb, hsproj, h_s, Wv, bv, ctx);
    build_x_kernel<<<BATCH, 256, 0, stream>>>(ctx, emb, tok, xb);
    // gi = x @ W_ih^T + b_ih ; gh = h @ W_hh^T + b_hh
    gemm(xb,    Wih_b, gi, b_ih, BATCH, H3, H2,  H3);
    gemm(hst_b, Whh_b, gh, b_hh, BATCH, H3, HID, H3);
    gru_gates_kernel<<<(BATCH * HID + 255) / 256, 256, 0, stream>>>(gi, gh, hst, hst_b);
    // logits = h_new @ Wout^T + bout  (dominant GEMM: 32x32000x512)
    gemm(hst_b, Wout_b, logits, bout, BATCH, VOCAB, HID, VOCAB);
    argmax_kernel<<<BATCH, 256, 0, stream>>>(logits, tok);
  }
}
